// TPQwenAttention_39470749450834
// MI455X (gfx1250) — matrix operand order repaired and checked by reading, not compiled, not run on hardware
//
#include <hip/hip_runtime.h>
#include <hip/hip_bf16.h>
#include <stdint.h>

#define HIDDEN   2048
#define N_HEADS  16
#define N_KVH    4
#define HEAD_DIM 128
#define KV_SIZE  512
#define BATCH    2
#define SEQ      2048

typedef __attribute__((ext_vector_type(16))) __bf16 v16bf;
typedef __attribute__((ext_vector_type(8)))  float  v8f;
typedef __attribute__((__vector_size__(4 * sizeof(int)))) int v4i_t;
typedef unsigned short ushort_t;

union FragAB { v16bf v; unsigned int u[8]; };

__device__ __forceinline__ unsigned int pack_bf16(float a, float b) {
  return (__float_as_uint(a) >> 16) | (__float_as_uint(b) & 0xFFFF0000u);
}
__device__ __forceinline__ ushort_t to_bf16(float a) {
  return (ushort_t)(__float_as_uint(a) >> 16);
}

// ---- CDNA5 async global->LDS copy (ASYNCcnt path), guarded fallback --------
__device__ __forceinline__ void async_copy16(const void* gsrc, void* ldst) {
#if defined(__has_builtin) && __has_builtin(__builtin_amdgcn_global_load_async_to_lds_b128)
  __builtin_amdgcn_global_load_async_to_lds_b128(
      (__attribute__((address_space(1))) v4i_t*)gsrc,
      (__attribute__((address_space(3))) v4i_t*)ldst, 0, 0);
#else
  *(uint4*)ldst = *(const uint4*)gsrc;
#endif
}
template<int N>
__device__ __forceinline__ void async_waitN() {
#if defined(__has_builtin) && __has_builtin(__builtin_amdgcn_s_wait_asynccnt)
  __builtin_amdgcn_s_wait_asynccnt(N);
#else
  asm volatile("s_wait_asynccnt 0x0" ::: "memory");
#endif
}

#define WMMA_BF16(A, B, Cacc) \
  __builtin_amdgcn_wmma_f32_16x16x32_bf16(false, (A).v, false, (B).v, (short)0, (Cacc), false, false)

// ---------------------------------------------------------------------------
// Elementwise f32 -> bf16 (n divisible by 2048)
// ---------------------------------------------------------------------------
__global__ __launch_bounds__(256)
void f32_to_bf16(const float* __restrict__ in, ushort_t* __restrict__ out) {
  long long i = ((long long)blockIdx.x * 256 + threadIdx.x) * 8;
  float4 a = ((const float4*)(in + i))[0];
  float4 b = ((const float4*)(in + i))[1];
  uint4 o;
  o.x = pack_bf16(a.x, a.y);
  o.y = pack_bf16(a.z, a.w);
  o.z = pack_bf16(b.x, b.y);
  o.w = pack_bf16(b.z, b.w);
  *(uint4*)(out + i) = o;
}

// ---------------------------------------------------------------------------
// NT GEMM, bf16 A and bf16 B in global memory:
//   C[m,n] = sum_k A[m,k] * B[n,k] (+ bias[n])
// Block = 256 thr = 8 waves; block tile 64(M) x 128(N); wave tile 32x32
// (4 WMMA / K-step, 2x fragment reuse). Double-buffered LDS, software-
// pipelined async global->LDS copies (wait asynccnt<=3 keeps next tile
// in flight while computing the current one).
// grid.z -> (b,h); B operand uses h/repB (GQA).
// CBF: store bf16 (else f32).  CT: store transposed C[n][m] (ldc = M-stride).
// ---------------------------------------------------------------------------
template<bool CBF, bool CT>
__global__ __launch_bounds__(256)
void gemm_nt_bf16(const ushort_t* __restrict__ A, long long lda, long long sAb, long long sAh,
                  const ushort_t* __restrict__ B, long long ldb, long long sBb, long long sBh, int repB,
                  void* __restrict__ C, long long ldc, long long sCb, long long sCh,
                  const float* __restrict__ bias, int K, int NH) {
  __shared__ unsigned int As[2][64 * 16];    // 2 x (64  rows x 32 bf16)
  __shared__ unsigned int Bs[2][128 * 16];   // 2 x (128 rows x 32 bf16)

  const int z = blockIdx.z;
  const int b = z / NH, h = z % NH;
  const ushort_t* Ab = A + (long long)b * sAb + (long long)h * sAh;
  const ushort_t* Bb = B + (long long)b * sBb + (long long)(h / repB) * sBh;

  const int tid = threadIdx.x, lane = tid & 31, wid = tid >> 5;
  const int m0 = blockIdx.x * 64, n0 = blockIdx.y * 128;
  const int wm = (wid >> 2) * 32, wn = (wid & 3) * 32;
  const int r = lane & 15, half = lane >> 4;

  const int lrow = tid >> 2, lchk = tid & 3;   // cooperative loader: 16B chunks

  v8f acc00 = {0.f,0.f,0.f,0.f,0.f,0.f,0.f,0.f};
  v8f acc01 = acc00, acc10 = acc00, acc11 = acc00;

  auto issue_tile = [&](int kb, int buf) {
    async_copy16(Ab + (long long)(m0 + lrow) * lda + kb + lchk * 8,
                 (char*)&As[buf][0] + lrow * 64 + lchk * 16);
    async_copy16(Bb + (long long)(n0 + lrow) * ldb + kb + lchk * 8,
                 (char*)&Bs[buf][0] + lrow * 64 + lchk * 16);
    async_copy16(Bb + (long long)(n0 + 64 + lrow) * ldb + kb + lchk * 8,
                 (char*)&Bs[buf][0] + (64 + lrow) * 64 + lchk * 16);
  };

  const int T = K >> 5;
  issue_tile(0, 0);
  for (int t = 0; t < T; ++t) {
    const int buf = t & 1;
    if (t + 1 < T) {                 // prefetch next tile into other buffer
      issue_tile((t + 1) << 5, buf ^ 1);
      async_waitN<3>();              // current tile's 3 copies complete
    } else {
      async_waitN<0>();
    }
    __syncthreads();

    FragAB fa0, fa1, fb0, fb1;
#pragma unroll
    for (int i = 0; i < 8; ++i) {
      // A 16x32 layout: vgpr i -> K = (i<4?0:16) + half*8 + 2*(i&3)
      int koff = ((i < 4) ? 0 : 8) + half * 4 + (i & 3);
      fa0.u[i] = As[buf][(wm + r) * 16 + koff];
      fa1.u[i] = As[buf][(wm + 16 + r) * 16 + koff];
      // B 32x16 layout: same K mapping as A
      fb0.u[i] = Bs[buf][(wn + r) * 16 + koff];
      fb1.u[i] = Bs[buf][(wn + 16 + r) * 16 + koff];
    }
    acc00 = WMMA_BF16(fa0, fb0, acc00);
    acc01 = WMMA_BF16(fa0, fb1, acc01);
    acc10 = WMMA_BF16(fa1, fb0, acc10);
    acc11 = WMMA_BF16(fa1, fb1, acc11);
    __syncthreads();                 // reads done; buffer reusable next+1 iter
  }

  const int cn0 = n0 + wn + r, cn1 = cn0 + 16;
  const float bv0 = bias ? bias[cn0] : 0.0f;
  const float bv1 = bias ? bias[cn1] : 0.0f;
  if (CBF) {
    ushort_t* Cb = (ushort_t*)C + (long long)b * sCb + (long long)h * sCh;
#pragma unroll
    for (int i = 0; i < 8; ++i) {
      long long cm0 = m0 + wm + i + half * 8, cm1 = cm0 + 16;
      if (CT) {  // transposed store: C[n][m]
        Cb[cn0 * ldc + cm0] = to_bf16(acc00[i] + bv0);
        Cb[cn1 * ldc + cm0] = to_bf16(acc01[i] + bv1);
        Cb[cn0 * ldc + cm1] = to_bf16(acc10[i] + bv0);
        Cb[cn1 * ldc + cm1] = to_bf16(acc11[i] + bv1);
      } else {
        Cb[cm0 * ldc + cn0] = to_bf16(acc00[i] + bv0);
        Cb[cm0 * ldc + cn1] = to_bf16(acc01[i] + bv1);
        Cb[cm1 * ldc + cn0] = to_bf16(acc10[i] + bv0);
        Cb[cm1 * ldc + cn1] = to_bf16(acc11[i] + bv1);
      }
    }
  } else {
    float* Cb = (float*)C + (long long)b * sCb + (long long)h * sCh;
#pragma unroll
    for (int i = 0; i < 8; ++i) {
      long long cm0 = m0 + wm + i + half * 8, cm1 = cm0 + 16;
      Cb[cm0 * ldc + cn0] = acc00[i] + bv0;
      Cb[cm0 * ldc + cn1] = acc01[i] + bv1;
      Cb[cm1 * ldc + cn0] = acc10[i] + bv0;
      Cb[cm1 * ldc + cn1] = acc11[i] + bv1;
    }
  }
}

// ---------------------------------------------------------------------------
// Row softmax over S=2048: p = softmax(p*scale + mask[q,:]) in place (fp32),
// plus a bf16 shadow copy for the downstream attn@V WMMA GEMM.
// ---------------------------------------------------------------------------
__global__ __launch_bounds__(256)
void softmax_rows(float* __restrict__ P, ushort_t* __restrict__ Pbf,
                  const float* __restrict__ mask, float scale) {
  __shared__ float red[256];
  const long long row = blockIdx.x;
  const int q = (int)(row % SEQ);
  float* p = P + row * (long long)SEQ;
  ushort_t* pb = Pbf + row * (long long)SEQ;
  const float* mrow = mask + (long long)q * SEQ;
  const int tid = threadIdx.x;

  float vals[8];
  float m = -3.4e38f;
#pragma unroll
  for (int i = 0; i < 8; ++i) {
    int k = tid + i * 256;
    float s = p[k] * scale + mrow[k];
    vals[i] = s;
    m = fmaxf(m, s);
  }
  red[tid] = m; __syncthreads();
  for (int off = 128; off > 0; off >>= 1) {
    if (tid < off) red[tid] = fmaxf(red[tid], red[tid + off]);
    __syncthreads();
  }
  m = red[0]; __syncthreads();

  float sum = 0.f;
#pragma unroll
  for (int i = 0; i < 8; ++i) { vals[i] = __expf(vals[i] - m); sum += vals[i]; }
  red[tid] = sum; __syncthreads();
  for (int off = 128; off > 0; off >>= 1) {
    if (tid < off) red[tid] += red[tid + off];
    __syncthreads();
  }
  float inv = 1.0f / red[0];
#pragma unroll
  for (int i = 0; i < 8; ++i) {
    float v = vals[i] * inv;
    p[tid + i * 256]  = v;
    pb[tid + i * 256] = to_bf16(v);
  }
}

// ---------------------------------------------------------------------------
extern "C" void kernel_launch(void* const* d_in, const int* in_sizes, int n_in,
                              void* d_out, int out_size, void* d_ws, size_t ws_size,
                              hipStream_t stream) {
  const float* x    = (const float*)d_in[0];
  const float* mask = (const float*)d_in[1];
  const float* wq   = (const float*)d_in[2];
  const float* bq   = (const float*)d_in[3];
  const float* wk   = (const float*)d_in[4];
  const float* bk   = (const float*)d_in[5];
  const float* wv   = (const float*)d_in[6];
  const float* bv   = (const float*)d_in[7];
  const float* wo   = (const float*)d_in[8];

  float* out  = (float*)d_out;                          // [B,S,HIDDEN]
  float* attn = out + (long long)BATCH * SEQ * HIDDEN;  // [B,H,S,S] fp32 (output)

  const long long M = (long long)BATCH * SEQ;  // 4096
  ushort_t* x_bf  = (ushort_t*)d_ws;                       // [M, HIDDEN]
  ushort_t* wq_bf = x_bf  + M * HIDDEN;                    // [HIDDEN, HIDDEN]
  ushort_t* wk_bf = wq_bf + (long long)HIDDEN * HIDDEN;    // [KV, HIDDEN]
  ushort_t* wv_bf = wk_bf + (long long)KV_SIZE * HIDDEN;
  ushort_t* wo_bf = wv_bf + (long long)KV_SIZE * HIDDEN;   // [HIDDEN, HIDDEN]
  ushort_t* q_bf  = wo_bf + (long long)HIDDEN * HIDDEN;    // [M, HIDDEN]
  ushort_t* k_bf  = q_bf  + M * HIDDEN;                    // [M, KV]
  ushort_t* vT_bf = k_bf  + M * KV_SIZE;                   // [KV, M] (transposed V)
  ushort_t* ao_bf = vT_bf + M * KV_SIZE;                   // [M, HIDDEN]
  ushort_t* p_bf  = ao_bf + M * HIDDEN;                    // [B,H,S,S] bf16 probs

  dim3 blk(256, 1, 1);
  // one-time f32 -> bf16 conversions
  f32_to_bf16<<<(unsigned)(M * HIDDEN / 2048), blk, 0, stream>>>(x,  x_bf);
  f32_to_bf16<<<(unsigned)((long long)HIDDEN * HIDDEN / 2048), blk, 0, stream>>>(wq, wq_bf);
  f32_to_bf16<<<(unsigned)((long long)KV_SIZE * HIDDEN / 2048), blk, 0, stream>>>(wk, wk_bf);
  f32_to_bf16<<<(unsigned)((long long)KV_SIZE * HIDDEN / 2048), blk, 0, stream>>>(wv, wv_bf);
  f32_to_bf16<<<(unsigned)((long long)HIDDEN * HIDDEN / 2048), blk, 0, stream>>>(wo, wo_bf);

  // Q/K projections (bf16 in, bf16 out, +bias); V projection stores transposed
  gemm_nt_bf16<true, false><<<dim3(M / 64, HIDDEN / 128, 1), blk, 0, stream>>>(
      x_bf, HIDDEN, 0, 0, wq_bf, HIDDEN, 0, 0, 1,
      q_bf, HIDDEN, 0, 0, bq, HIDDEN, 1);
  gemm_nt_bf16<true, false><<<dim3(M / 64, KV_SIZE / 128, 1), blk, 0, stream>>>(
      x_bf, HIDDEN, 0, 0, wk_bf, HIDDEN, 0, 0, 1,
      k_bf, KV_SIZE, 0, 0, bk, HIDDEN, 1);
  gemm_nt_bf16<true, true><<<dim3(M / 64, KV_SIZE / 128, 1), blk, 0, stream>>>(
      x_bf, HIDDEN, 0, 0, wv_bf, HIDDEN, 0, 0, 1,
      vT_bf, M /*ldc: n-major*/, 0, 0, bv, HIDDEN, 1);

  // scores[b,h] = Q[b,h] @ K[b,h/4]^T -> fp32 attn region (scale+mask in softmax)
  gemm_nt_bf16<false, false><<<dim3(SEQ / 64, SEQ / 128, BATCH * N_HEADS), blk, 0, stream>>>(
      q_bf, HIDDEN, (long long)SEQ * HIDDEN, HEAD_DIM,
      k_bf, KV_SIZE, (long long)SEQ * KV_SIZE, HEAD_DIM, N_HEADS / N_KVH,
      attn, SEQ, (long long)N_HEADS * SEQ * SEQ, (long long)SEQ * SEQ,
      nullptr, HEAD_DIM, N_HEADS);

  // softmax(p/sqrt(128) + causal_mask): fp32 in place (output) + bf16 shadow
  softmax_rows<<<dim3(BATCH * N_HEADS * SEQ, 1, 1), blk, 0, stream>>>(
      attn, p_bf, mask, 0.08838834764831845f);

  // attn_out[b,h] = P_bf[b,h] @ vT[b,h/4]^T  (same NT kernel: O[q,d]=sum_k P[q,k]*vT[d,k])
  gemm_nt_bf16<true, false><<<dim3(SEQ / 64, HEAD_DIM / 128, BATCH * N_HEADS), blk, 0, stream>>>(
      p_bf, SEQ, (long long)N_HEADS * SEQ * SEQ, (long long)SEQ * SEQ,
      vT_bf, M, (long long)SEQ /*b: column block*/, (long long)HEAD_DIM * M, N_HEADS / N_KVH,
      ao_bf, HIDDEN, (long long)SEQ * HIDDEN, HEAD_DIM,
      nullptr, SEQ, N_HEADS);

  // out = attn_out @ wo^T (fp32 out, no bias)
  gemm_nt_bf16<false, false><<<dim3(M / 64, HIDDEN / 128, 1), blk, 0, stream>>>(
      ao_bf, HIDDEN, 0, 0, wo_bf, HIDDEN, 0, 0, 1,
      out, HIDDEN, 0, 0, nullptr, HIDDEN, 1);
}
